// LSTMModel_1958505087487
// MI455X (gfx1250) — compile-verified, matrix operand order checked by reading, not run on hardware
//
#include <hip/hip_runtime.h>
#include <hip/hip_bf16.h>

// ---------------------------------------------------------------------------
// LSTM: B=128, T=2048, D=128, H=256 (4H = 1024), bf16 WMMA on gfx1250.
// Phase 1: xg[t][g][b] = x[b,t,:]·w_x2h[g,:] + b_x2h[g] + b_h2h[g]  (parallel)
// Phase 2: persistent 16-WG kernel, per-step grid barrier, h double-buffered,
//          w_h2h slice resident in LDS (staged via TDM), c-state in VGPRs.
// Phase 3: out[b] = sigmoid(h_T[b,:]·w_fc + b_fc)
// ---------------------------------------------------------------------------

typedef __bf16 bf16;
typedef __attribute__((ext_vector_type(16))) __bf16 v16bf;
typedef __attribute__((ext_vector_type(8)))  __bf16 v8bf;
typedef __attribute__((ext_vector_type(8)))  float  v8f;

union FragAB { v16bf v; uint4 q[2]; };   // 32 B: one 16x32 bf16 A/B fragment
union Half8  { v8bf  v; uint4 q;    };   // 16 B: 8 bf16

constexpr int BB = 128;    // batch
constexpr int TT = 2048;   // timesteps
constexpr int DD = 128;    // input dim
constexpr int HH = 256;    // hidden
constexpr int GG = 1024;   // 4H
constexpr int NWG = 16;    // workgroups in recurrent phase (16 h-cols each)

// Workspace layout (bytes). Everything 16B-aligned.
constexpr size_t XG_OFF = 0;
constexpr size_t XG_SZ  = (size_t)TT * GG * BB * 2;   // 512 MiB, bf16 [t][g][b]
constexpr size_t XB_OFF = XG_OFF + XG_SZ;
constexpr size_t XB_SZ  = (size_t)BB * TT * DD * 2;   // 64 MiB, bf16 x
constexpr size_t WX_OFF = XB_OFF + XB_SZ;
constexpr size_t WX_SZ  = (size_t)GG * DD * 2;
constexpr size_t WH_OFF = WX_OFF + WX_SZ;
constexpr size_t WH_SZ  = (size_t)GG * HH * 2;
constexpr size_t BI_OFF = WH_OFF + WH_SZ;
constexpr size_t BI_SZ  = (size_t)GG * 4;
constexpr size_t H0_OFF = BI_OFF + BI_SZ;
constexpr size_t H_SZ   = (size_t)BB * HH * 2;
constexpr size_t H1_OFF = H0_OFF + H_SZ;
constexpr size_t CNT_OFF = H1_OFF + H_SZ;

__device__ __forceinline__ float fsig(float x) {
    return 1.0f / (1.0f + __expf(-x));
}
__device__ __forceinline__ float ftanh(float x) {
    x = fminf(15.0f, fmaxf(-15.0f, x));
    float e = __expf(2.0f * x);
    return (e - 1.0f) / (e + 1.0f);
}

// ---------------------------------------------------------------------------
// Tensor Data Mover helpers (D# per CDNA5 ISA §8). 2D tile, bf16 elements,
// optional LDS row padding. Groups 2/3 zero (<=2D tensor, fields unused).
// ---------------------------------------------------------------------------
#if defined(__AMDGCN__) && __has_builtin(__builtin_amdgcn_tensor_load_to_lds) && \
    __has_builtin(__builtin_amdgcn_s_wait_tensorcnt)
#define LSTM_HAS_TDM 1
typedef __attribute__((ext_vector_type(4))) unsigned tdm_v4u;
typedef __attribute__((ext_vector_type(8))) int      tdm_v8i;
typedef __attribute__((ext_vector_type(4))) int      tdm_v4i;

// tile_d0: contiguous elements per row; tile_d1: rows; row_stride: elements
// pad_int_code: LDS pad interval code (2^(c+1) DWORDs); pad_amt_code: (c+1) DWORDs
__device__ __forceinline__ void tdm_load_2d_bf16(unsigned lds_off, const bf16* gptr,
                                                 unsigned tile_d0, unsigned tile_d1,
                                                 unsigned long long row_stride,
                                                 unsigned pad_int_code,
                                                 unsigned pad_amt_code) {
    unsigned long long ga = (unsigned long long)(uintptr_t)gptr;
    tdm_v4u g0;
    g0[0] = 1u;                                              // count=1 (valid)
    g0[1] = lds_off;                                         // lds_addr (bytes)
    g0[2] = (unsigned)(ga & 0xFFFFFFFFull);                  // global_addr[31:0]
    g0[3] = (unsigned)((ga >> 32) & 0x01FFFFFFull) | (2u << 30);  // [56:32] | type=2

    tdm_v8i g1;
    g1[0] = (int)((1u << 16) |                               // data_size = 2 B
                  (1u << 20) |                               // pad_enable
                  (pad_int_code << 22) | (pad_amt_code << 25));
    g1[1] = (int)((tile_d0 & 0xFFFFu) << 16);                // tensor_dim0[15:0]
    g1[2] = (int)(((tile_d0 >> 16) & 0xFFFFu) |              // tensor_dim0[31:16]
                  ((tile_d1 & 0xFFFFu) << 16));              // tensor_dim1[15:0]
    g1[3] = (int)(((tile_d1 >> 16) & 0xFFFFu) |              // tensor_dim1[31:16]
                  ((tile_d0 & 0xFFFFu) << 16));              // tile_dim0
    g1[4] = (int)(tile_d1 & 0xFFFFu);                        // tile_dim1 (tile_dim2=0)
    g1[5] = (int)(row_stride & 0xFFFFFFFFull);               // tensor_dim0_stride[31:0]
    g1[6] = (int)((row_stride >> 32) & 0xFFFFull);           // stride[47:32], dim1_stride lo = 0
    g1[7] = 0;

    tdm_v4i gz4 = {0, 0, 0, 0};
#if __clang_major__ >= 23
    tdm_v8i gz8 = {0, 0, 0, 0, 0, 0, 0, 0};
    __builtin_amdgcn_tensor_load_to_lds(g0, g1, gz4, gz4, gz8, 0);
#else
    __builtin_amdgcn_tensor_load_to_lds(g0, g1, gz4, gz4, 0);
#endif
}
#else
#define LSTM_HAS_TDM 0
#endif

// ---------------------------------------------------------------------------
// Prep: fp32->bf16 conversions, bias sum, zero h buffers & sync counter.
// ---------------------------------------------------------------------------
__global__ void lstm_prep(const float* __restrict__ x,
                          const float* __restrict__ wx,
                          const float* __restrict__ bx,
                          const float* __restrict__ wh,
                          const float* __restrict__ bh,
                          bf16* __restrict__ xb, bf16* __restrict__ wxb,
                          bf16* __restrict__ whb, float* __restrict__ bias,
                          bf16* __restrict__ h0, bf16* __restrict__ h1,
                          unsigned* __restrict__ cnt) {
    size_t tid = (size_t)blockIdx.x * blockDim.x + threadIdx.x;
    size_t nth = (size_t)gridDim.x * blockDim.x;
    for (size_t i = tid; i < (size_t)BB * TT * DD; i += nth) xb[i] = (bf16)x[i];
    for (size_t i = tid; i < (size_t)GG * DD; i += nth)      wxb[i] = (bf16)wx[i];
    for (size_t i = tid; i < (size_t)GG * HH; i += nth)      whb[i] = (bf16)wh[i];
    for (size_t i = tid; i < (size_t)GG; i += nth)           bias[i] = bx[i] + bh[i];
    for (size_t i = tid; i < (size_t)BB * HH; i += nth) {
        h0[i] = (bf16)0.0f;
        h1[i] = (bf16)0.0f;
    }
    if (tid == 0) *cnt = 0u;
}

// ---------------------------------------------------------------------------
// Phase 1: xg[t][g][b] = wx(g,:)·x(b,t,:) + bias[g].  One block per t.
// Output C tile layout: M = g (rows), N = b (cols).
// ---------------------------------------------------------------------------
__global__ __launch_bounds__(256) void lstm_xproj(const bf16* __restrict__ xb,
                                                  const bf16* __restrict__ wxb,
                                                  const float* __restrict__ bias,
                                                  bf16* __restrict__ xg) {
    const int t = blockIdx.x;
    const int lane = threadIdx.x & 31, wave = threadIdx.x >> 5;
    const int half = lane >> 4, l16 = lane & 15;
    const int k0base = half * 8;

    __shared__ bf16 xl[128 * 136];   // x_t staged [b][d], padded stride 136
#if LSTM_HAS_TDM
    // TDM: 128 rows x 128 bf16, global row stride T*D, LDS pad 256B->272B
    // (row = 64 DWORDs -> interval code 5; pad = 4 DWORDs -> amount code 3)
    if (threadIdx.x < 32) {
        tdm_load_2d_bf16((unsigned)(uintptr_t)&xl[0], xb + (size_t)t * DD,
                         128u, 128u, (unsigned long long)TT * DD, 5u, 3u);
        __builtin_amdgcn_s_wait_tensorcnt(0);
    }
#else
    for (int idx = threadIdx.x; idx < 128 * 128; idx += 256) {
        int b = idx >> 7, d = idx & 127;
        xl[b * 136 + d] = xb[((size_t)b * TT + t) * DD + d];
    }
#endif
    __syncthreads();

    for (int mm = 0; mm < 8; ++mm) {
        const int mt = wave * 8 + mm;          // g-tile 0..63
        const int arow = mt * 16 + l16;
        FragAB afr[4];
#pragma unroll
        for (int kk = 0; kk < 4; ++kk) {       // K = 128 -> 4 x 32
            const bf16* ap = wxb + (size_t)arow * DD + kk * 32 + k0base;
            afr[kk].q[0] = *(const uint4*)ap;
            afr[kk].q[1] = *(const uint4*)(ap + 16);
        }
        float binit[8];
#pragma unroll
        for (int v = 0; v < 8; ++v) binit[v] = bias[mt * 16 + v + half * 8];

        for (int nt = 0; nt < 8; ++nt) {       // b-tile 0..7
            v8f acc;
#pragma unroll
            for (int v = 0; v < 8; ++v) acc[v] = binit[v];
#pragma unroll
            for (int kk = 0; kk < 4; ++kk) {
                FragAB bfr;
                const bf16* bp = &xl[(nt * 16 + l16) * 136 + kk * 32 + k0base];
                bfr.q[0] = *(const uint4*)bp;
                bfr.q[1] = *(const uint4*)(bp + 16);
                acc = __builtin_amdgcn_wmma_f32_16x16x32_bf16(
                    false, afr[kk].v, false, bfr.v, (short)0, acc, false, false);
            }
            const int bcol = nt * 16 + l16;
#pragma unroll
            for (int v = 0; v < 8; ++v) {
                int g = mt * 16 + v + half * 8;
                xg[((size_t)t * GG + g) * BB + bcol] = (bf16)acc[v];
            }
        }
    }
}

// ---------------------------------------------------------------------------
// Phase 2: persistent recurrent kernel. 16 WGs x 256 threads (8 waves).
// WG wg owns h columns [wg*16, wg*16+16) -> gate cols {s*256 + wg*16 + j}.
// Wave w owns batch rows [w*16, w*16+16). c-state lives in VGPRs.
// ---------------------------------------------------------------------------
__global__ __launch_bounds__(256) void lstm_recurrent(const bf16* __restrict__ xg,
                                                      const bf16* __restrict__ whb,
                                                      bf16* __restrict__ h0,
                                                      bf16* __restrict__ h1,
                                                      unsigned* __restrict__ cnt) {
    const int wg = blockIdx.x;                    // 0..15
    const int lane = threadIdx.x & 31, wave = threadIdx.x >> 5;
    const int half = lane >> 4, l16 = lane & 15;
    const int k0base = half * 8;

    // Resident w_h2h slice: 64 gate rows x 256 k, padded stride 272 (16B-mult).
    __shared__ bf16 whl[64 * 272];
#if LSTM_HAS_TDM
    // TDM: 4 strips of 16 rows x 256 bf16; global row stride 256 elements;
    // LDS pad 512B->544B (row = 128 DWORDs -> code 6; pad = 8 DWORDs -> code 7)
    if (threadIdx.x < 32) {
#pragma unroll
        for (int s = 0; s < 4; ++s) {
            tdm_load_2d_bf16((unsigned)(uintptr_t)&whl[s * 16 * 272],
                             whb + (size_t)(s * 256 + wg * 16) * HH,
                             256u, 16u, (unsigned long long)HH, 6u, 7u);
        }
        __builtin_amdgcn_s_wait_tensorcnt(0);
    }
#else
    for (int idx = threadIdx.x; idx < 64 * 256; idx += 256) {
        int r = idx >> 8, k = idx & 255;
        int g = (r >> 4) * 256 + wg * 16 + (r & 15);
        whl[r * 272 + k] = whb[(size_t)g * HH + k];
    }
#endif
    __syncthreads();

    float c[8];
#pragma unroll
    for (int v = 0; v < 8; ++v) c[v] = 0.0f;

    bf16* hbuf[2] = { h0, h1 };
    const int hcol = wg * 16 + l16;
    const int b0 = wave * 16 + half * 8;

    for (int t = 0; t < TT; ++t) {
        const bf16* __restrict__ hr = hbuf[t & 1];
        bf16* __restrict__ hw = hbuf[(t & 1) ^ 1];

        // C init = xg tile (already includes both biases), one b128 per strip.
        v8f acc[4];
        const size_t xgt = (size_t)t * GG;
#pragma unroll
        for (int s = 0; s < 4; ++s) {
            int g = s * 256 + wg * 16 + l16;
            Half8 x8;
            x8.q = *(const uint4*)(xg + (xgt + g) * BB + b0);
#pragma unroll
            for (int v = 0; v < 8; ++v) acc[s][v] = (float)x8.v[v];
        }
        if (t + 1 < TT) {   // global_prefetch next step's xg line
            __builtin_prefetch(xg + ((size_t)(t + 1) * GG + wg * 16 + l16) * BB + b0, 0, 1);
        }

        // gates += h @ w_h2h^T  (K = 256 -> 8 x 32)
#pragma unroll
        for (int kk = 0; kk < 8; ++kk) {
            FragAB afr;
            const bf16* ap = hr + (size_t)(wave * 16 + l16) * HH + kk * 32 + k0base;
            afr.q[0] = *(const uint4*)ap;
            afr.q[1] = *(const uint4*)(ap + 16);
#pragma unroll
            for (int s = 0; s < 4; ++s) {
                FragAB bfr;
                const bf16* bp = &whl[(s * 16 + l16) * 272 + kk * 32 + k0base];
                bfr.q[0] = *(const uint4*)bp;
                bfr.q[1] = *(const uint4*)(bp + 16);
                acc[s] = __builtin_amdgcn_wmma_f32_16x16x32_bf16(
                    false, afr.v, false, bfr.v, (short)0, acc[s], false, false);
            }
        }

        // LSTM cell update; i/f/g/o for (b, hcol) are lane-local.
#pragma unroll
        for (int v = 0; v < 8; ++v) {
            float ig = fsig(acc[0][v]);
            float fg = fsig(acc[1][v]);
            float gg = ftanh(acc[2][v]);
            float og = fsig(acc[3][v]);
            c[v] = c[v] * fg + ig * gg;
            float hv = og * ftanh(c[v]);
            hw[(size_t)(wave * 16 + v + half * 8) * HH + hcol] = (bf16)hv;
        }

        // device-wide step barrier (monotonic counter)
        __threadfence();
        __syncthreads();
        if (threadIdx.x == 0) {
            __hip_atomic_fetch_add(cnt, 1u, __ATOMIC_RELEASE, __HIP_MEMORY_SCOPE_AGENT);
            const unsigned target = (unsigned)(t + 1) * (unsigned)NWG;
            while (__hip_atomic_load(cnt, __ATOMIC_ACQUIRE, __HIP_MEMORY_SCOPE_AGENT) < target) {
                __builtin_amdgcn_s_sleep(2);
            }
        }
        __syncthreads();
        __threadfence();
    }
}

// ---------------------------------------------------------------------------
// Phase 3: out[b] = sigmoid(h_T[b,:]·w_fc + b_fc). Final h is in hbuf[0].
// ---------------------------------------------------------------------------
__global__ void lstm_fc(const bf16* __restrict__ hfin,
                        const float* __restrict__ wfc,
                        const float* __restrict__ bfc,
                        float* __restrict__ out) {
    int b = threadIdx.x;
    if (b < BB) {
        float accv = bfc[0];
#pragma unroll 8
        for (int k = 0; k < HH; ++k) accv += (float)hfin[(size_t)b * HH + k] * wfc[k];
        out[b] = fsig(accv);
    }
}

// ---------------------------------------------------------------------------
extern "C" void kernel_launch(void* const* d_in, const int* in_sizes, int n_in,
                              void* d_out, int out_size, void* d_ws, size_t ws_size,
                              hipStream_t stream) {
    const float* x   = (const float*)d_in[0];  // [B, T, D]
    const float* wx  = (const float*)d_in[1];  // [4H, D]
    const float* bx  = (const float*)d_in[2];  // [4H]
    const float* wh  = (const float*)d_in[3];  // [4H, H]
    const float* bh  = (const float*)d_in[4];  // [4H]
    const float* wfc = (const float*)d_in[5];  // [1, H]
    const float* bfc = (const float*)d_in[6];  // [1]
    float* out = (float*)d_out;

    char* ws = (char*)d_ws;
    bf16*     xg   = (bf16*)(ws + XG_OFF);
    bf16*     xb   = (bf16*)(ws + XB_OFF);
    bf16*     wxb  = (bf16*)(ws + WX_OFF);
    bf16*     whb  = (bf16*)(ws + WH_OFF);
    float*    bias = (float*)(ws + BI_OFF);
    bf16*     h0   = (bf16*)(ws + H0_OFF);
    bf16*     h1   = (bf16*)(ws + H1_OFF);
    unsigned* cnt  = (unsigned*)(ws + CNT_OFF);

    lstm_prep<<<2048, 256, 0, stream>>>(x, wx, bx, wh, bh, xb, wxb, whb, bias, h0, h1, cnt);
    lstm_xproj<<<TT, 256, 0, stream>>>(xb, wxb, bias, xg);
    lstm_recurrent<<<NWG, 256, 0, stream>>>(xg, whb, h0, h1, cnt);
    lstm_fc<<<1, 128, 0, stream>>>(h0, wfc, bfc, out);
}
